// CausalSelfAttention_24026047054506
// MI455X (gfx1250) — matrix operand order repaired and checked by reading, not compiled, not run on hardware
//
#include <hip/hip_runtime.h>
#include <hip/hip_bf16.h>

// ---------------------------------------------------------------------------
// CDNA5 (gfx1250) causal self-attention: bf16 WMMA + async-to-LDS staging.
//   B=4, T=2048, C=1024, H=16, Dh=64
// Pipeline:
//   0) one-shot f32->bf16 packs: xb[M][K], wqkvT[N][K], wprojT[N][K]
//   1) qkv GEMM (async double-buffered LDS, 64x64 wave tiles) -> Q,K,V^T bf16
//   2) flash attention (16 q-rows/wave, shuffle softmax, LDS P-bounce)
//   3) proj GEMM -> f32 out
// ---------------------------------------------------------------------------

typedef __attribute__((ext_vector_type(16))) __bf16        v16bf;
typedef __attribute__((ext_vector_type(8)))  float         v8f;
typedef __attribute__((ext_vector_type(4)))  unsigned int  u32x4;

union Frag {
    v16bf v;
    u32x4 q[2];
    unsigned int u[8];
};

// round-to-nearest-even f32 -> bf16
__device__ __forceinline__ unsigned int pack_bf16(float a, float b) {
    unsigned int ua = __builtin_bit_cast(unsigned int, a);
    unsigned int ub = __builtin_bit_cast(unsigned int, b);
    ua += 0x7fffu + ((ua >> 16) & 1u);
    ub += 0x7fffu + ((ub >> 16) & 1u);
    return (ua >> 16) | (ub & 0xffff0000u);
}
__device__ __forceinline__ unsigned short bf16_1(float a) {
    unsigned int ua = __builtin_bit_cast(unsigned int, a);
    ua += 0x7fffu + ((ua >> 16) & 1u);
    return (unsigned short)(ua >> 16);
}

__device__ __forceinline__ v8f wmma_bf16(const Frag& a, const Frag& b, v8f c) {
    return __builtin_amdgcn_wmma_f32_16x16x32_bf16(
        false, a.v, false, b.v, (short)0, c, false, false);
}

// async copy 16B/lane global -> LDS (ASYNCcnt-tracked, no VGPR data)
__device__ __forceinline__ void async_b128(unsigned lds_off, const void* gaddr) {
    asm volatile("global_load_async_to_lds_b128 %0, %1, off"
                 :: "v"(lds_off), "v"((unsigned long long)(size_t)gaddr)
                 : "memory");
}

// ---------------------------------------------------------------------------
// Kernel 0a: pack f32 pairs -> bf16 pairs (row-major copy)
// ---------------------------------------------------------------------------
__global__ __launch_bounds__(256) void cvt_pack_kernel(
    const float* __restrict__ src, unsigned int* __restrict__ dst, int npairs)
{
    int i = blockIdx.x * 256 + threadIdx.x;
    if (i < npairs) dst[i] = pack_bf16(src[2 * i], src[2 * i + 1]);
}

// ---------------------------------------------------------------------------
// Kernel 0b: transpose + pack: w[K][N] f32 -> wT[N][K/2] bf16-pair uints
// ---------------------------------------------------------------------------
__global__ __launch_bounds__(256) void cvt_transpose_kernel(
    const float* __restrict__ w, unsigned int* __restrict__ wT, int N)
{
    // u = kp * N + n  (consecutive tid -> consecutive n: coalesced reads)
    unsigned int u = blockIdx.x * 256 + threadIdx.x;
    int n  = (int)(u % (unsigned)N);
    int kp = (int)(u / (unsigned)N);
    if (kp < 512) {   // K = 1024 for both weights
        const float* p = w + (size_t)(2 * kp) * N + n;
        wT[(size_t)n * 512 + kp] = pack_bf16(p[0], p[N]);
    }
}

// ---------------------------------------------------------------------------
// Shared GEMM machinery: C[M][N] (f32 frags) = Ab[M][K] * BtT[N][K], K=1024.
// Block tile 256(M) x 128(N), wave tile 64x64, K-step 32, double-buffered
// async-to-LDS staging (6 x b128 per thread per tile).
// ---------------------------------------------------------------------------
#define GEMM_KSTEPS 32   // 1024 / 32

__device__ __forceinline__ void stage_tile_async(
    const unsigned int* __restrict__ Ab, const unsigned int* __restrict__ Bt,
    int m0, int n0, int kp0, int tid,
    unsigned int (*__restrict__ As)[16], unsigned int (*__restrict__ Bs)[16])
{
    // A: 256 rows x 4 uint4 (row = 32 bf16 of this k-slice)
    #pragma unroll
    for (int i = 0; i < 4; i++) {
        int u = i * 256 + tid;             // 0..1023
        int row = u >> 2, c = u & 3;
        const void* g = Ab + (size_t)(m0 + row) * 512 + kp0 + c * 4;
        async_b128((unsigned)(size_t)&As[row][c * 4], g);
    }
    // B: 128 rows x 4 uint4
    #pragma unroll
    for (int i = 0; i < 2; i++) {
        int u = i * 256 + tid;             // 0..511
        int row = u >> 2, c = u & 3;
        const void* g = Bt + (size_t)(n0 + row) * 512 + kp0 + c * 4;
        async_b128((unsigned)(size_t)&Bs[row][c * 4], g);
    }
}

struct GemmAcc { v8f acc[4][4]; };

__device__ __forceinline__ void gemm_core(
    const unsigned int* __restrict__ Ab,   // bf16 pairs [M][512]
    const unsigned int* __restrict__ Bt,   // bf16 pairs [N][512]
    int m0, int n0, int tid,
    unsigned int (*__restrict__ As)[256][16],   // [2][256][16]
    unsigned int (*__restrict__ Bs)[128][16],   // [2][128][16]
    GemmAcc& r)
{
    const int lane = tid & 31;
    const int wv   = tid >> 5;
    const int wm   = (wv & 3) * 64;    // 4 waves along M
    const int wn   = (wv >> 2) * 64;   // 2 waves along N
    const int half = lane >> 4;
    const int l16  = lane & 15;

    #pragma unroll
    for (int i = 0; i < 4; i++)
        #pragma unroll
        for (int j = 0; j < 4; j++)
            #pragma unroll
            for (int e = 0; e < 8; e++) r.acc[i][j][e] = 0.f;

    stage_tile_async(Ab, Bt, m0, n0, 0, tid, As[0], Bs[0]);

    for (int kc = 0; kc < GEMM_KSTEPS; kc++) {
        if (kc + 1 < GEMM_KSTEPS) {
            stage_tile_async(Ab, Bt, m0, n0, (kc + 1) * 16, tid,
                             As[(kc + 1) & 1], Bs[(kc + 1) & 1]);
            asm volatile("s_wait_asynccnt 6" ::: "memory");  // prev tile done
        } else {
            asm volatile("s_wait_asynccnt 0" ::: "memory");
        }
        __syncthreads();   // LDS visible to all waves

        unsigned int (*Ac)[16] = As[kc & 1];
        unsigned int (*Bc)[16] = Bs[kc & 1];

        Frag a[4], b[4];
        #pragma unroll
        for (int mi = 0; mi < 4; mi++) {
            int row = wm + mi * 16 + l16;
            a[mi].q[0] = *(const u32x4*)&Ac[row][half * 4];      // k 0..7 / 8..15
            a[mi].q[1] = *(const u32x4*)&Ac[row][8 + half * 4];  // k 16..23 / 24..31
        }
        #pragma unroll
        for (int ni = 0; ni < 4; ni++) {
            int col = wn + ni * 16 + l16;
            b[ni].q[0] = *(const u32x4*)&Bc[col][half * 4];
            b[ni].q[1] = *(const u32x4*)&Bc[col][8 + half * 4];
        }
        #pragma unroll
        for (int mi = 0; mi < 4; mi++)
            #pragma unroll
            for (int ni = 0; ni < 4; ni++)
                r.acc[mi][ni] = wmma_bf16(a[mi], b[ni], r.acc[mi][ni]);

        __syncthreads();   // WAR guard before next-tile overwrite
    }
}

// ---------------------------------------------------------------------------
// Kernel 1: qkv = xb @ wqkvT^T  (M=8192, N=3072) -> scatter Q,K (bf16,
// [B,H,T,64]) and V transposed ([B,H,64,T]) for attention B-fragments.
// ---------------------------------------------------------------------------
__global__ __launch_bounds__(256) void qkv_gemm_kernel(
    const unsigned int* __restrict__ xb,      // [8192][512] bf16 pairs
    const unsigned int* __restrict__ wqkvT,   // [3072][512] bf16 pairs
    unsigned short* __restrict__ Qb,
    unsigned short* __restrict__ Kb,
    unsigned short* __restrict__ Vt)
{
    __shared__ __align__(16) unsigned int As[2][256][16];
    __shared__ __align__(16) unsigned int Bs[2][128][16];

    const int tid  = threadIdx.x;
    const int lane = tid & 31;
    const int wv   = tid >> 5;
    const int m0   = blockIdx.y * 256;
    const int n0   = blockIdx.x * 128;
    const int wm   = (wv & 3) * 64;
    const int wn   = (wv >> 2) * 64;
    const int half = lane >> 4;
    const int l16  = lane & 15;

    GemmAcc r;
    gemm_core(xb, wqkvT, m0, n0, tid, As, Bs, r);

    #pragma unroll
    for (int mi = 0; mi < 4; mi++) {
        #pragma unroll
        for (int ni = 0; ni < 4; ni++) {
            #pragma unroll
            for (int rr = 0; rr < 8; rr++) {
                int m = m0 + wm + mi * 16 + rr + half * 8;
                int n = n0 + wn + ni * 16 + l16;
                int bb = m >> 11, t = m & 2047;
                int which = n >> 10, c = n & 1023;
                int h = c >> 6, d = c & 63;
                int bh = bb * 16 + h;
                unsigned short val = bf16_1(r.acc[mi][ni][rr]);
                if (which == 0)
                    Qb[((size_t)bh * 2048 + t) * 64 + d] = val;
                else if (which == 1)
                    Kb[((size_t)bh * 2048 + t) * 64 + d] = val;
                else
                    Vt[((size_t)bh * 64 + d) * 2048 + t] = val;
            }
        }
    }
}

// ---------------------------------------------------------------------------
// Kernel 2: flash attention (unchanged structure; 8 WMMA per 32-key chunk)
// ---------------------------------------------------------------------------
__global__ __launch_bounds__(256) void flash_attn_kernel(
    const unsigned short* __restrict__ Qb,   // [64][2048][64]
    const unsigned short* __restrict__ Kb,   // [64][2048][64]
    const unsigned short* __restrict__ Vt,   // [64][64][2048]
    unsigned short* __restrict__ O)          // [8192][1024] bf16
{
    __shared__ __align__(16) unsigned short ps[8][16][32];  // per-wave P scratch

    const int tid  = threadIdx.x;
    const int lane = tid & 31;
    const int wv   = tid >> 5;
    const int bh   = blockIdx.x >> 4;
    const int qt   = blockIdx.x & 15;
    const int q0   = qt * 128 + wv * 16;
    const int half = lane >> 4;
    const int l16  = lane & 15;
    const int h    = bh & 15;
    const int bb   = bh >> 4;

    const unsigned short* Qrow = Qb + ((size_t)bh * 2048 + q0 + l16) * 64;
    Frag qa[2];
    qa[0].q[0] = *(const u32x4*)(Qrow + half * 8);
    qa[0].q[1] = *(const u32x4*)(Qrow + 16 + half * 8);
    qa[1].q[0] = *(const u32x4*)(Qrow + 32 + half * 8);
    qa[1].q[1] = *(const u32x4*)(Qrow + 48 + half * 8);

    v8f o[4];
    #pragma unroll
    for (int f = 0; f < 4; f++)
        #pragma unroll
        for (int e = 0; e < 8; e++) o[f][e] = 0.f;

    float mrow[8], lrow[8];
    #pragma unroll
    for (int rr = 0; rr < 8; rr++) { mrow[rr] = -INFINITY; lrow[rr] = 0.f; }

    const float scale = 0.125f;
    const int nchunk = ((q0 + 15) >> 5) + 1;

    for (int kc = 0; kc < nchunk; kc++) {
        const int k0 = kc * 32;

        Frag kb[2][2];
        #pragma unroll
        for (int ni = 0; ni < 2; ni++) {
            const unsigned short* Krow =
                Kb + ((size_t)bh * 2048 + k0 + ni * 16 + l16) * 64;
            kb[ni][0].q[0] = *(const u32x4*)(Krow + half * 8);
            kb[ni][0].q[1] = *(const u32x4*)(Krow + 16 + half * 8);
            kb[ni][1].q[0] = *(const u32x4*)(Krow + 32 + half * 8);
            kb[ni][1].q[1] = *(const u32x4*)(Krow + 48 + half * 8);
        }
        Frag vb[4];
        #pragma unroll
        for (int f = 0; f < 4; f++) {
            const unsigned short* Vrow =
                Vt + ((size_t)bh * 64 + f * 16 + l16) * 2048 + k0 + half * 16;
            vb[f].q[0] = *(const u32x4*)(Vrow);
            vb[f].q[1] = *(const u32x4*)(Vrow + 8);
        }

        v8f zero;
        #pragma unroll
        for (int e = 0; e < 8; e++) zero[e] = 0.f;
        v8f s0 = wmma_bf16(qa[0], kb[0][0], zero);
        s0     = wmma_bf16(qa[1], kb[0][1], s0);
        v8f s1 = wmma_bf16(qa[0], kb[1][0], zero);
        s1     = wmma_bf16(qa[1], kb[1][1], s1);

        float p0[8], p1[8], alpha[8];
        #pragma unroll
        for (int rr = 0; rr < 8; rr++) {
            const int q = q0 + rr + half * 8;
            float v0 = s0[rr] * scale;
            float v1 = s1[rr] * scale;
            if (k0 + l16 > q)      v0 = -1e30f;
            if (k0 + 16 + l16 > q) v1 = -1e30f;
            float mx = fmaxf(v0, v1);
            mx = fmaxf(mx, __shfl_xor(mx, 1));
            mx = fmaxf(mx, __shfl_xor(mx, 2));
            mx = fmaxf(mx, __shfl_xor(mx, 4));
            mx = fmaxf(mx, __shfl_xor(mx, 8));
            const float mnew = fmaxf(mrow[rr], mx);
            alpha[rr] = __expf(mrow[rr] - mnew);
            mrow[rr]  = mnew;
            p0[rr] = __expf(v0 - mnew);
            p1[rr] = __expf(v1 - mnew);
            lrow[rr] = lrow[rr] * alpha[rr] + p0[rr] + p1[rr];
        }
        #pragma unroll
        for (int f = 0; f < 4; f++)
            #pragma unroll
            for (int rr = 0; rr < 8; rr++) o[f][rr] *= alpha[rr];

        #pragma unroll
        for (int rr = 0; rr < 8; rr++) {
            const int row = rr + half * 8;
            ps[wv][row][l16]      = bf16_1(p0[rr]);
            ps[wv][row][16 + l16] = bf16_1(p1[rr]);
        }
        asm volatile("s_wait_dscnt 0" ::: "memory");  // DS in-order, wave-local
        Frag pA;
        pA.q[0] = *(const u32x4*)&ps[wv][l16][half * 16];
        pA.q[1] = *(const u32x4*)&ps[wv][l16][half * 16 + 8];

        #pragma unroll
        for (int f = 0; f < 4; f++) o[f] = wmma_bf16(pA, vb[f], o[f]);
    }

    float linv[8];
    #pragma unroll
    for (int rr = 0; rr < 8; rr++) {
        float l = lrow[rr];
        l += __shfl_xor(l, 1);
        l += __shfl_xor(l, 2);
        l += __shfl_xor(l, 4);
        l += __shfl_xor(l, 8);
        linv[rr] = 1.f / l;
    }
    #pragma unroll
    for (int f = 0; f < 4; f++) {
        #pragma unroll
        for (int rr = 0; rr < 8; rr++) {
            const int q = q0 + rr + half * 8;
            const size_t idx =
                ((size_t)(bb * 2048 + q)) * 1024 + h * 64 + f * 16 + l16;
            O[idx] = bf16_1(o[f][rr] * linv[rr]);
        }
    }
}

// ---------------------------------------------------------------------------
// Kernel 3: out(f32) = AO(bf16) @ wprojT^T  (M=8192, N=1024)
// ---------------------------------------------------------------------------
__global__ __launch_bounds__(256) void proj_gemm_kernel(
    const unsigned int* __restrict__ AO,      // [8192][512] bf16 pairs
    const unsigned int* __restrict__ wprojT,  // [1024][512] bf16 pairs
    float* __restrict__ out)                  // [8192][1024] f32
{
    __shared__ __align__(16) unsigned int As[2][256][16];
    __shared__ __align__(16) unsigned int Bs[2][128][16];

    const int tid  = threadIdx.x;
    const int lane = tid & 31;
    const int wv   = tid >> 5;
    const int m0   = blockIdx.y * 256;
    const int n0   = blockIdx.x * 128;
    const int wm   = (wv & 3) * 64;
    const int wn   = (wv >> 2) * 64;
    const int half = lane >> 4;
    const int l16  = lane & 15;

    GemmAcc r;
    gemm_core(AO, wprojT, m0, n0, tid, As, Bs, r);

    #pragma unroll
    for (int mi = 0; mi < 4; mi++) {
        #pragma unroll
        for (int ni = 0; ni < 4; ni++) {
            #pragma unroll
            for (int rr = 0; rr < 8; rr++) {
                int m = m0 + wm + mi * 16 + rr + half * 8;
                int n = n0 + wn + ni * 16 + l16;
                out[(size_t)m * 1024 + n] = r.acc[mi][ni][rr];
            }
        }
    }
}

// ---------------------------------------------------------------------------
// Launch
// ---------------------------------------------------------------------------
extern "C" void kernel_launch(void* const* d_in, const int* in_sizes, int n_in,
                              void* d_out, int out_size, void* d_ws, size_t ws_size,
                              hipStream_t stream) {
    const float* x      = (const float*)d_in[0];   // [4,2048,1024]
    const float* w_qkv  = (const float*)d_in[1];   // [1024,3072]
    const float* w_proj = (const float*)d_in[2];   // [1024,1024]
    float* out = (float*)d_out;

    // workspace carve-up (bf16 elements)
    unsigned short* p = (unsigned short*)d_ws;
    const size_t XB   = (size_t)8192 * 1024;       // xb
    const size_t WQT  = (size_t)3072 * 1024;       // wqkvT
    const size_t WPT  = (size_t)1024 * 1024;       // wprojT
    const size_t HEAD = (size_t)64 * 2048 * 64;    // per Q/K/Vt
    unsigned short* xb     = p;              p += XB;
    unsigned short* wqkvT  = p;              p += WQT;
    unsigned short* wprojT = p;              p += WPT;
    unsigned short* Qb     = p;              p += HEAD;
    unsigned short* Kb     = p;              p += HEAD;
    unsigned short* Vt     = p;              p += HEAD;
    unsigned short* AO     = p;              p += XB;

    dim3 blk(256);

    // 0) one-shot bf16 packs / transposed weight packs
    cvt_pack_kernel<<<dim3(16384), blk, 0, stream>>>(
        x, (unsigned int*)xb, 8192 * 512);
    cvt_transpose_kernel<<<dim3(6144), blk, 0, stream>>>(
        w_qkv, (unsigned int*)wqkvT, 3072);
    cvt_transpose_kernel<<<dim3(2048), blk, 0, stream>>>(
        w_proj, (unsigned int*)wprojT, 1024);

    // 1) QKV projection
    qkv_gemm_kernel<<<dim3(24, 32), blk, 0, stream>>>(
        (const unsigned int*)xb, (const unsigned int*)wqkvT, Qb, Kb, Vt);

    // 2) flash attention
    flash_attn_kernel<<<dim3(1024), blk, 0, stream>>>(Qb, Kb, Vt, AO);

    // 3) output projection
    proj_gemm_kernel<<<dim3(8, 32), blk, 0, stream>>>(
        (const unsigned int*)AO, (const unsigned int*)wprojT, out);
}
